// PiscesModel_27023934226607
// MI455X (gfx1250) — matrix operand order repaired and checked by reading, not compiled, not run on hardware
//
#include <hip/hip_runtime.h>
#include <cstddef>
#include <cstdint>

// ---------------- model constants ----------------
constexpr int Lnum = 4, Dm = 1024, Hn = 16, KVH = 4, HD = 64;
constexpr int En = 8, TopK = 2, Fm = 512;
constexpr int Vv = 32000, Bb = 2, Tt = 1024;
constexpr int NTOK = Bb * Tt;                 // 2048
constexpr int TASK = 10, EVALN = 5;
constexpr size_t OUT_TASK = (size_t)Bb * Tt * Vv;      // 65,536,000
constexpr size_t OUT_EVAL = OUT_TASK + Bb * TASK;      // +20
constexpr size_t OUT_AUX  = OUT_EVAL + Bb * EVALN;     // +10

// ---------------- WMMA types ----------------
typedef __attribute__((ext_vector_type(16))) __bf16 v16bf;
typedef __attribute__((ext_vector_type(8)))  __bf16 v8bf;
typedef __attribute__((ext_vector_type(8)))  float  v8f;
typedef __attribute__((ext_vector_type(4)))  int    v4i_t;

__device__ __forceinline__ unsigned short f2bf(float f) {
  unsigned int u = __float_as_uint(f);
  unsigned int r = (u + 0x7FFFu + ((u >> 16) & 1u)) >> 16;   // RNE
  return (unsigned short)r;
}

// ---- async global->LDS staging (gfx1250 ASYNCcnt DMA path), with fallback ----
#if __has_builtin(__builtin_amdgcn_global_load_async_to_lds_b128)
  // builtin signature (from compiler diagnostics): (v4i AS1*, v4i AS3*, Ii offset, Ii cpol)
  #define ASYNC_LDS_COPY16(gp, lp)                                             \
    __builtin_amdgcn_global_load_async_to_lds_b128(                            \
        (v4i_t __attribute__((address_space(1)))*)(unsigned long long)(gp),    \
        (v4i_t __attribute__((address_space(3)))*)(unsigned int)               \
            (unsigned long long)(lp),                                          \
        0, 0)
  #if __has_builtin(__builtin_amdgcn_s_wait_asynccnt)
    #define ASYNC_LDS_WAIT() __builtin_amdgcn_s_wait_asynccnt(0)
  #else
    #define ASYNC_LDS_WAIT() asm volatile("s_wait_asynccnt 0x0" ::: "memory")
  #endif
#else
  #define ASYNC_LDS_COPY16(gp, lp) (*(float4*)(lp) = *(const float4*)(gp))
  #define ASYNC_LDS_WAIT() ((void)0)
  #warning "no async-to-LDS builtin available; sync staging fallback"
#endif

// ---------------- generic GEMM: C[M,N] = A[M,K](bf16) @ B[K,N](f32), f32 accum --------
// requires K % 64 == 0; M,N guarded (all call sites use M=2048, K in {512,1024}).
#define BM 128
#define BN 64
#define BKK 64   // two WMMA K-chunks of 32 per LDS stage

__global__ __launch_bounds__(256) void gemm_bf16_kernel(
    const unsigned short* __restrict__ A, const float* __restrict__ Bm,
    float* __restrict__ C, int M, int N, int Kd)
{
  __shared__ alignas(64) unsigned short As[BM * BKK];   // 16 KB, row-major [m][k]
  __shared__ alignas(64) unsigned short Bs[BN * BKK];   //  8 KB, transposed [n][k]
  const int tid  = threadIdx.x;
  const int lane = tid & 31;
  const int wave = tid >> 5;
  const int m0 = blockIdx.y * BM;
  const int n0 = blockIdx.x * BN;

  v8f zero;
  #pragma unroll
  for (int i = 0; i < 8; ++i) zero[i] = 0.0f;
  v8f acc[4] = { zero, zero, zero, zero };

  for (int k0 = 0; k0 < Kd; k0 += BKK) {
    // ---- stage A tile (128x64 bf16) via async LDS DMA: pure copy, no VALU ----
    {
      const int r = tid >> 1;                 // 128 rows / 2 threads per row
      const int c = (tid & 1) * 32;           // 32 bf16 = 64 B per thread
      const int gm = m0 + r;
      if (gm < M) {
        const unsigned short* gp = A + (size_t)gm * Kd + k0 + c;
        #pragma unroll
        for (int i = 0; i < 4; ++i)
          ASYNC_LDS_COPY16(gp + 8 * i, &As[r * BKK + c + 8 * i]);
      } else {
        #pragma unroll
        for (int i = 0; i < 32; ++i) As[r * BKK + c + i] = 0;
      }
    }
    // ---- stage B tile (64x64 f32 -> bf16, transposed [n][k]) ----
    {
      const int kk = tid >> 2;                // 64 k-rows
      const int nb = (tid & 3) * 16;          // 16 n-cols per thread
      #pragma unroll
      for (int i = 0; i < 4; ++i) {
        const int gn = n0 + nb + 4 * i;
        float4 v = make_float4(0.f, 0.f, 0.f, 0.f);
        if (gn + 3 < N) {
          v = *(const float4*)(Bm + (size_t)(k0 + kk) * N + gn);
          if (k0 + BKK < Kd)
            __builtin_prefetch(Bm + (size_t)(k0 + BKK + kk) * N + gn, 0, 1);
        } else {
          if (gn + 0 < N) v.x = Bm[(size_t)(k0 + kk) * N + gn + 0];
          if (gn + 1 < N) v.y = Bm[(size_t)(k0 + kk) * N + gn + 1];
          if (gn + 2 < N) v.z = Bm[(size_t)(k0 + kk) * N + gn + 2];
          if (gn + 3 < N) v.w = Bm[(size_t)(k0 + kk) * N + gn + 3];
        }
        Bs[(nb + 4 * i + 0) * BKK + kk] = f2bf(v.x);
        Bs[(nb + 4 * i + 1) * BKK + kk] = f2bf(v.y);
        Bs[(nb + 4 * i + 2) * BKK + kk] = f2bf(v.z);
        Bs[(nb + 4 * i + 3) * BKK + kk] = f2bf(v.w);
      }
    }
    ASYNC_LDS_WAIT();
    __syncthreads();

    const int am = wave * 16 + (lane & 15);
    const int s  = lane >> 4;
    #pragma unroll
    for (int kc = 0; kc < BKK; kc += 32) {
      // A fragment (contiguous K order, matching B: K = 16s..16s+15)
      v16bf afrag = *(const v16bf*)&As[am * BKK + kc + 16 * s];
      // 4 B fragments + WMMA
      #pragma unroll
      for (int j = 0; j < 4; ++j) {
        const int bn = j * 16 + (lane & 15);
        v16bf bfrag = *(const v16bf*)&Bs[bn * BKK + kc + 16 * s];
        acc[j] = __builtin_amdgcn_wmma_f32_16x16x32_bf16(
            false, afrag, false, bfrag, (short)0, acc[j], false, false);
      }
    }
    __syncthreads();
  }

  // ---- store C (ISA f32 C/D layout) ----
  const int mrow  = m0 + wave * 16 + (lane >> 4) * 8;
  const int ncol0 = n0 + (lane & 15);
  #pragma unroll
  for (int j = 0; j < 4; ++j) {
    const int nc = ncol0 + j * 16;
    if (nc < N) {
      #pragma unroll
      for (int r = 0; r < 8; ++r) {
        const int gm = mrow + r;
        if (gm < M) C[(size_t)gm * N + nc] = acc[j][r];
      }
    }
  }
}

// ---------------- elementwise / support kernels ----------------
__global__ void embed_kernel(const int* __restrict__ ids, const float* __restrict__ ew,
                             float* __restrict__ x) {
  const int t = blockIdx.x;
  const int id = ids[t];
  for (int d = threadIdx.x; d < Dm; d += blockDim.x)
    x[(size_t)t * Dm + d] = ew[(size_t)id * Dm + d];
}

// writes f32 (for heads) and bf16 (GEMM A operand)
__global__ __launch_bounds__(256) void rmsnorm_kernel(const float* __restrict__ x,
                                                      const float* __restrict__ w,
                                                      float* __restrict__ out,
                                                      unsigned short* __restrict__ outb) {
  __shared__ float red[256];
  const int t = blockIdx.x;
  const float* xr = x + (size_t)t * Dm;
  float s = 0.f;
  for (int d = threadIdx.x; d < Dm; d += 256) { float v = xr[d]; s += v * v; }
  red[threadIdx.x] = s;
  __syncthreads();
  for (int off = 128; off > 0; off >>= 1) {
    if (threadIdx.x < off) red[threadIdx.x] += red[threadIdx.x + off];
    __syncthreads();
  }
  const float rs = rsqrtf(red[0] / Dm + 1e-6f);
  for (int d = threadIdx.x; d < Dm; d += 256) {
    const float v = w[d] * xr[d] * rs;
    out[(size_t)t * Dm + d] = v;
    outb[(size_t)t * Dm + d] = f2bf(v);
  }
}

__global__ void rope_kernel(float* __restrict__ p, int nheads, int stride) {
  const int tok = blockIdx.x;
  const int t = tok % Tt;
  const int h = threadIdx.x >> 5;      // 32 pairs per head
  const int i = threadIdx.x & 31;
  if (h >= nheads) return;
  const float inv = __expf(-(2.0f * i / 64.0f) * 13.815510558f); // ln(1e6)
  const float fr = (float)t * inv;
  const float c = __cosf(fr), sn = __sinf(fr);
  float* q = p + (size_t)tok * stride + h * HD + 2 * i;
  const float x0 = q[0], x1 = q[1];
  q[0] = x0 * c - x1 * sn;
  q[1] = x0 * sn + x1 * c;
}

// one block = 64 query rows of one (b,h); streaming online softmax over causal keys
__global__ __launch_bounds__(64) void attention_kernel(const float* __restrict__ q,
                                                       const float* __restrict__ k,
                                                       const float* __restrict__ v,
                                                       unsigned short* __restrict__ outb) {
  __shared__ float ks[64 * 64];
  __shared__ float vs[64 * 64];
  const int bh = blockIdx.y;
  const int b = bh / Hn, h = bh % Hn, kh = h / (Hn / KVH);
  const int qi = blockIdx.x * 64 + threadIdx.x;          // query index in [0,T)
  const int tok = b * Tt + qi;

  float qreg[64], o[64];
  #pragma unroll
  for (int d = 0; d < 64; ++d) {
    qreg[d] = q[(size_t)tok * (Hn * HD) + h * HD + d];
    o[d] = 0.f;
  }
  float mrun = -3.0e38f, lsum = 0.f;

  for (int kt = 0; kt <= (int)blockIdx.x; ++kt) {
    {
      const int row = b * Tt + kt * 64 + threadIdx.x;
      const float4* kp = (const float4*)(k + (size_t)row * (KVH * HD) + kh * HD);
      const float4* vp = (const float4*)(v + (size_t)row * (KVH * HD) + kh * HD);
      #pragma unroll
      for (int d4 = 0; d4 < 16; ++d4) {
        ((float4*)&ks[threadIdx.x * 64])[d4] = kp[d4];
        ((float4*)&vs[threadIdx.x * 64])[d4] = vp[d4];
      }
    }
    __syncthreads();
    #pragma unroll 1
    for (int jj = 0; jj < 64; ++jj) {
      const int j = kt * 64 + jj;
      if (j <= qi) {
        float s = 0.f;
        #pragma unroll
        for (int d = 0; d < 64; ++d) s += qreg[d] * ks[jj * 64 + d];
        s *= 0.125f;                                    // HD^-0.5
        const float nm = fmaxf(mrun, s);
        const float corr = __expf(mrun - nm);
        const float pe = __expf(s - nm);
        lsum = lsum * corr + pe;
        #pragma unroll
        for (int d = 0; d < 64; ++d) o[d] = o[d] * corr + pe * vs[jj * 64 + d];
        mrun = nm;
      }
    }
    __syncthreads();
  }
  const float rl = 1.0f / lsum;
  #pragma unroll
  for (int d = 0; d < 64; ++d)
    outb[(size_t)tok * (Hn * HD) + h * HD + d] = f2bf(o[d] * rl);
}

__global__ void router_kernel(const float* __restrict__ rl, float* __restrict__ gates,
                              float* __restrict__ me, float* __restrict__ cnt) {
  const int i = blockIdx.x * blockDim.x + threadIdx.x;
  if (i >= NTOK) return;
  float p[En];
  float mx = -3.0e38f;
  #pragma unroll
  for (int e = 0; e < En; ++e) { p[e] = rl[i * En + e]; mx = fmaxf(mx, p[e]); }
  float sum = 0.f;
  #pragma unroll
  for (int e = 0; e < En; ++e) { p[e] = __expf(p[e] - mx); sum += p[e]; }
  const float rs = 1.0f / sum;
  #pragma unroll
  for (int e = 0; e < En; ++e) p[e] *= rs;
  int i0 = 0;
  #pragma unroll
  for (int e = 1; e < En; ++e) if (p[e] > p[i0]) i0 = e;
  int i1 = (i0 == 0) ? 1 : 0;
  #pragma unroll
  for (int e = 0; e < En; ++e) if (e != i0 && p[e] > p[i1]) i1 = e;
  const float g = p[i0] + p[i1];
  #pragma unroll
  for (int e = 0; e < En; ++e) gates[i * En + e] = 0.f;
  gates[i * En + i0] = p[i0] / g;
  gates[i * En + i1] = p[i1] / g;
  #pragma unroll
  for (int e = 0; e < En; ++e) atomicAdd(&me[e], p[e]);
  atomicAdd(&cnt[i0], 1.0f);
  atomicAdd(&cnt[i1], 1.0f);
}

__global__ void aux_kernel(const float* __restrict__ me, const float* __restrict__ cnt,
                           float* __restrict__ auxout) {
  if (threadIdx.x == 0 && blockIdx.x == 0) {
    float a = 0.f;
    for (int e = 0; e < En; ++e)
      a += (me[e] / NTOK) * (cnt[e] / (float)(NTOK * TopK));
    auxout[0] += (float)En * a;
  }
}

__global__ void silu_mul_kernel(const float* __restrict__ hg, const float* __restrict__ hu,
                                unsigned short* __restrict__ outb, int n) {
  const int i = blockIdx.x * blockDim.x + threadIdx.x;
  if (i < n) {
    const float g = hg[i];
    outb[i] = f2bf((g / (1.0f + __expf(-g))) * hu[i]);
  }
}

__global__ void moe_accum_kernel(float* __restrict__ out, const float* __restrict__ y,
                                 const float* __restrict__ gates, int e) {
  const int i = blockIdx.x;
  const float g = gates[i * En + e];
  for (int d = threadIdx.x; d < Dm; d += blockDim.x)
    out[(size_t)i * Dm + d] += g * y[(size_t)i * Dm + d];
}

__global__ void add_inplace_kernel(float* __restrict__ dst, const float* __restrict__ src, int n) {
  const int i = blockIdx.x * blockDim.x + threadIdx.x;
  if (i < n) dst[i] += src[i];
}

__global__ void xmean_kernel(const float* __restrict__ xn, float* __restrict__ xm) {
  const int idx = blockIdx.x * blockDim.x + threadIdx.x;
  if (idx >= Bb * Dm) return;
  const int b = idx / Dm, d = idx % Dm;
  float s = 0.f;
  for (int t = 0; t < Tt; ++t) s += xn[((size_t)(b * Tt + t)) * Dm + d];
  xm[idx] = s / (float)Tt;
}

__global__ void heads_kernel(const float* __restrict__ xn, const float* __restrict__ xm,
                             const float* __restrict__ tw, const float* __restrict__ tb,
                             const float* __restrict__ ew, const float* __restrict__ eb,
                             float* __restrict__ out_task, float* __restrict__ out_eval) {
  const int tid = threadIdx.x;
  if (tid < Bb * TASK) {
    const int b = tid / TASK, j = tid % TASK;
    const float* xr = xn + (size_t)(b * Tt) * Dm;   // x[:,0]
    float s = tb[j];
    for (int d = 0; d < Dm; ++d) s += xr[d] * tw[d * TASK + j];
    out_task[b * TASK + j] = s;
  } else if (tid >= 32 && tid < 32 + Bb * EVALN) {
    const int u = tid - 32;
    const int b = u / EVALN, j = u % EVALN;
    float s = eb[j];
    for (int d = 0; d < Dm; ++d) s += xm[b * Dm + d] * ew[d * EVALN + j];
    out_eval[b * EVALN + j] = s;
  }
}

// ---------------- host launch ----------------
static inline void gemm(const unsigned short* A, const float* B, float* C,
                        int M, int N, int K, hipStream_t stream) {
  dim3 grid((N + BN - 1) / BN, (M + BM - 1) / BM);
  gemm_bf16_kernel<<<grid, 256, 0, stream>>>(A, B, C, M, N, K);
}

extern "C" void kernel_launch(void* const* d_in, const int* in_sizes, int n_in,
                              void* d_out, int out_size, void* d_ws, size_t ws_size,
                              hipStream_t stream) {
  const int*   ids      = (const int*)d_in[0];
  const float* embed_w  = (const float*)d_in[1];
  const float* wq       = (const float*)d_in[2];
  const float* wk       = (const float*)d_in[3];
  const float* wv       = (const float*)d_in[4];
  const float* wo       = (const float*)d_in[5];
  const float* norm1_w  = (const float*)d_in[6];
  const float* norm2_w  = (const float*)d_in[7];
  const float* router_w = (const float*)d_in[8];
  const float* w_gate   = (const float*)d_in[9];
  const float* w_up     = (const float*)d_in[10];
  const float* w_down   = (const float*)d_in[11];
  const float* normf_w  = (const float*)d_in[12];
  const float* lm_head  = (const float*)d_in[13];
  const float* task_w   = (const float*)d_in[14];
  const float* task_b   = (const float*)d_in[15];
  const float* eval_w   = (const float*)d_in[16];
  const float* eval_b   = (const float*)d_in[17];

  float* out = (float*)d_out;
  float* ws  = (float*)d_ws;

  // workspace partition (float-sized slots; bf16 buffers use half-slots)
  size_t o = 0;
  float* X    = ws + o; o += (size_t)NTOK * Dm;
  float* XN   = ws + o; o += (size_t)NTOK * Dm;
  float* Q    = ws + o; o += (size_t)NTOK * Hn * HD;
  float* Kb   = ws + o; o += (size_t)NTOK * KVH * HD;
  float* Vb   = ws + o; o += (size_t)NTOK * KVH * HD;
  float* TMP  = ws + o; o += (size_t)NTOK * Dm;
  float* HG   = ws + o; o += (size_t)NTOK * Fm;
  float* HU   = ws + o; o += (size_t)NTOK * Fm;
  float* RL   = ws + o; o += (size_t)NTOK * En;
  float* GT   = ws + o; o += (size_t)NTOK * En;
  float* MO   = ws + o; o += (size_t)NTOK * Dm;
  float* XM   = ws + o; o += (size_t)Bb * Dm;
  float* ME   = ws + o; o += En;
  float* CT   = ws + o; o += En;
  unsigned short* XNb = (unsigned short*)(ws + o); o += (size_t)NTOK * Dm / 2;
  unsigned short* AOb = (unsigned short*)(ws + o); o += (size_t)NTOK * Hn * HD / 2;
  unsigned short* HGb = (unsigned short*)(ws + o); o += (size_t)NTOK * Fm / 2;
  (void)ws_size; (void)in_sizes; (void)n_in; (void)out_size;

  (void)hipMemsetAsync(out + OUT_AUX, 0, sizeof(float), stream);

  embed_kernel<<<NTOK, 256, 0, stream>>>(ids, embed_w, X);

  for (int l = 0; l < Lnum; ++l) {
    // ---- attention block ----
    rmsnorm_kernel<<<NTOK, 256, 0, stream>>>(X, norm1_w + (size_t)l * Dm, XN, XNb);
    gemm(XNb, wq + (size_t)l * Dm * (Hn * HD), Q,  NTOK, Hn * HD,  Dm, stream);
    gemm(XNb, wk + (size_t)l * Dm * (KVH * HD), Kb, NTOK, KVH * HD, Dm, stream);
    gemm(XNb, wv + (size_t)l * Dm * (KVH * HD), Vb, NTOK, KVH * HD, Dm, stream);
    rope_kernel<<<NTOK, Hn * 32, 0, stream>>>(Q, Hn, Hn * HD);
    rope_kernel<<<NTOK, KVH * 32, 0, stream>>>(Kb, KVH, KVH * HD);
    attention_kernel<<<dim3(Tt / 64, Bb * Hn), 64, 0, stream>>>(Q, Kb, Vb, AOb);
    gemm(AOb, wo + (size_t)l * (Hn * HD) * Dm, TMP, NTOK, Dm, Hn * HD, stream);
    add_inplace_kernel<<<(NTOK * Dm + 255) / 256, 256, 0, stream>>>(X, TMP, NTOK * Dm);

    // ---- MoE block ----
    rmsnorm_kernel<<<NTOK, 256, 0, stream>>>(X, norm2_w + (size_t)l * Dm, XN, XNb);
    gemm(XNb, router_w + (size_t)l * Dm * En, RL, NTOK, En, Dm, stream);
    (void)hipMemsetAsync(ME, 0, 2 * En * sizeof(float), stream);   // ME + CT contiguous
    router_kernel<<<(NTOK + 255) / 256, 256, 0, stream>>>(RL, GT, ME, CT);
    aux_kernel<<<1, 32, 0, stream>>>(ME, CT, out + OUT_AUX);
    (void)hipMemsetAsync(MO, 0, (size_t)NTOK * Dm * sizeof(float), stream);
    for (int e = 0; e < En; ++e) {
      const size_t wgo = ((size_t)l * En + e) * Dm * Fm;
      const size_t wdo = ((size_t)l * En + e) * Fm * Dm;
      gemm(XNb, w_gate + wgo, HG, NTOK, Fm, Dm, stream);
      gemm(XNb, w_up + wgo,   HU, NTOK, Fm, Dm, stream);
      silu_mul_kernel<<<(NTOK * Fm + 255) / 256, 256, 0, stream>>>(HG, HU, HGb, NTOK * Fm);
      gemm(HGb, w_down + wdo, TMP, NTOK, Dm, Fm, stream);
      moe_accum_kernel<<<NTOK, 256, 0, stream>>>(MO, TMP, GT, e);
    }
    add_inplace_kernel<<<(NTOK * Dm + 255) / 256, 256, 0, stream>>>(X, MO, NTOK * Dm);
  }

  // ---- final norm + heads ----
  rmsnorm_kernel<<<NTOK, 256, 0, stream>>>(X, normf_w, XN, XNb);
  gemm(XNb, lm_head, out, NTOK, Vv, Dm, stream);                 // logits
  xmean_kernel<<<(Bb * Dm + 255) / 256, 256, 0, stream>>>(XN, XM);
  heads_kernel<<<1, 64, 0, stream>>>(XN, XM, task_w, task_b, eval_w, eval_b,
                                     out + OUT_TASK, out + OUT_EVAL);
}
